// NGP_50414326120788
// MI455X (gfx1250) — compile-verified, matrix operand order checked by reading, not run on hardware
//
#include <hip/hip_runtime.h>
#include <hip/hip_bf16.h>
#include <math.h>

// ---------------------------------------------------------------------------
// Fused Instant-NGP forward for MI455X (gfx1250, wave32, WMMA).
//  Phase A (1 point/lane): mask + occupancy + 16-level hash-grid trilinear
//    gather (L2-resident tables) -> f16 features in LDS.
//  Phase B (per-wave WMMA): 32->64->16 density MLP and 43(->64)->64->64->3
//    color MLP entirely in LDS, v_wmma_f32_16x16x32_f16, f32 accumulate.
// ---------------------------------------------------------------------------

typedef _Float16 f16;
typedef f16   v8h  __attribute__((ext_vector_type(8)));
typedef f16   v16h __attribute__((ext_vector_type(16)));
typedef float v8f  __attribute__((ext_vector_type(8)));

#define HASH_T    524288
#define HASH_MASK (HASH_T - 1)
#define NLEV      16
#define SRES      128

__constant__ int c_levels[NLEV] = {16, 20, 25, 32, 40, 50, 64, 80,
                                   101, 128, 161, 203, 256, 322, 406, 512};

// Load one 16-bit WMMA operand fragment (A or B^T) from an LDS row.
// CDNA5 16-bit operand layout: lanes 0-15 hold K = {0..7, 16..23} (+kbase),
// lanes 16-31 hold K = {8..15, 24..31}.  Two aligned 16-byte LDS loads.
static __device__ inline v16h frag_ld(const f16* row, int kbase, bool hi) {
  const f16* p = row + kbase + (hi ? 8 : 0);
  v8h lo = *(const v8h*)(p);
  v8h hh = *(const v8h*)(p + 16);
  return __builtin_shufflevector(lo, hh, 0, 1, 2, 3, 4, 5, 6, 7,
                                          8, 9, 10, 11, 12, 13, 14, 15);
}

static __device__ inline v8f wmma16(v16h a, v16h b, v8f c) {
  // D = A x B + C, f16 inputs, f32 accum (v_wmma_f32_16x16x32_f16)
  return __builtin_amdgcn_wmma_f32_16x16x32_f16(false, a, false, b,
                                                (short)0, c, false, false);
}

__global__ __launch_bounds__(128) void ngp_fused_kernel(
    const float* __restrict__ x, const float* __restrict__ rdir,
    const unsigned char* __restrict__ occ, const float* __restrict__ tables,
    const float* __restrict__ dW1, const float* __restrict__ db1,
    const float* __restrict__ dW2, const float* __restrict__ db2,
    const float* __restrict__ cW1, const float* __restrict__ cb1,
    const float* __restrict__ cW2, const float* __restrict__ cb2,
    const float* __restrict__ cW3, const float* __restrict__ cb3,
    float* __restrict__ outColor, float* __restrict__ outSigma,
    float* __restrict__ outMask, int P) {
  // Activations: 2 ping-pong buffers x 4 waves x 32 points x 64 f16.
  __shared__ alignas(16) f16 sAct[2][4][32][64];
  // Weights, stored TRANSPOSED ([N][K]) so B fragments are K-contiguous.
  __shared__ alignas(16) f16 sW1[64][32];   // density L1: K=32 -> N=64
  __shared__ alignas(16) f16 sW2[16][64];   // density L2: K=64 -> N=16
  __shared__ alignas(16) f16 sC1[64][64];   // color  L1: K=43(pad64) -> N=64
  __shared__ alignas(16) f16 sC2[64][64];   // color  L2: K=64 -> N=64
  __shared__ alignas(16) f16 sC3[16][64];   // color  L3: K=64 -> N=3(pad16)
  __shared__ float sB1[64], sB2[16], sCB1[64], sCB2[64], sCB3[3];
  __shared__ float sMask[128];

  const int tid  = threadIdx.x;
  const int wave = tid >> 5;
  const int lane = tid & 31;
  const bool hi  = lane >= 16;
  const int lc   = lane & 15;
  const int blockbase = blockIdx.x * 128;
  const int point = blockbase + tid;
  const int p = point < P ? point : (P - 1);

  // ---- cooperative weight load (f32 -> f16, transposed, zero-padded) ----
  for (int i = tid; i < 64 * 32; i += 128) {
    int n = i >> 5, k = i & 31;
    sW1[n][k] = (f16)dW1[k * 64 + n];
  }
  for (int i = tid; i < 16 * 64; i += 128) {
    int n = i >> 6, k = i & 63;
    sW2[n][k] = (f16)dW2[k * 16 + n];
  }
  for (int i = tid; i < 64 * 64; i += 128) {
    int n = i >> 6, k = i & 63;
    sC1[n][k] = (k < 43) ? (f16)cW1[k * 64 + n] : (f16)0.f;
    sC2[n][k] = (f16)cW2[k * 64 + n];
  }
  for (int i = tid; i < 16 * 64; i += 128) {
    int n = i >> 6, k = i & 63;
    sC3[n][k] = (n < 3) ? (f16)cW3[k * 3 + n] : (f16)0.f;
  }
  if (tid < 64) { sB1[tid] = db1[tid]; sCB1[tid] = cb1[tid]; sCB2[tid] = cb2[tid]; }
  if (tid < 16) sB2[tid] = db2[tid];
  if (tid < 3)  sCB3[tid] = cb3[tid];

  // ---- Phase A: mask + hash-grid features (one point per lane) ----
  float x0 = x[p * 3 + 0] * 0.5f + 0.5f;
  float x1 = x[p * 3 + 1] * 0.5f + 0.5f;
  float x2 = x[p * 3 + 2] * 0.5f + 0.5f;
  bool inb = (x0 > 0.f) & (x0 < 1.f) & (x1 > 0.f) & (x1 < 1.f) &
             (x2 > 0.f) & (x2 < 1.f);
  int i0 = (int)floorf(fminf(fmaxf(x0, 0.f), 0.99f) * (float)SRES);
  int i1 = (int)floorf(fminf(fmaxf(x1, 0.f), 0.99f) * (float)SRES);
  int i2 = (int)floorf(fminf(fmaxf(x2, 0.f), 0.99f) * (float)SRES);
  bool m = inb && (occ[(i0 * SRES + i1) * SRES + i2] != 0);
  float mf = m ? 1.f : 0.f;
  sMask[tid] = mf;
  if (point < P) outMask[point] = mf;
  float s0 = m ? x0 : 0.5f, s1 = m ? x1 : 0.5f, s2 = m ? x2 : 0.5f;

  f16* myrow = &sAct[0][wave][lane][0];
  for (int lv = 0; lv < NLEV; ++lv) {
    float lf = (float)c_levels[lv];
    float sx = s0 * lf, sy = s1 * lf, sz = s2 * lf;
    float fx = floorf(sx), fy = floorf(sy), fz = floorf(sz);
    float tx = sx - fx, ty = sy - fy, tz = sz - fz;
    unsigned bx = (unsigned)fx, by = (unsigned)fy, bz = (unsigned)fz;
    const float2* tab = (const float2*)(tables + (size_t)lv * (HASH_T * 2));
    float a0 = 0.f, a1 = 0.f;
#pragma unroll
    for (int c = 0; c < 8; ++c) {
      unsigned cx = bx + (c & 1), cy = by + ((c >> 1) & 1),
               cz = bz + ((c >> 2) & 1);
      unsigned h = (cx * 1u) ^ (cy * 2654435761u) ^ (cz * 805459861u);
      h &= HASH_MASK;
      float w = ((c & 1) ? tx : 1.f - tx) * (((c >> 1) & 1) ? ty : 1.f - ty) *
                (((c >> 2) & 1) ? tz : 1.f - tz);
      float2 f = tab[h];
      a0 += w * f.x;
      a1 += w * f.y;
    }
    myrow[2 * lv + 0] = (f16)a0;
    myrow[2 * lv + 1] = (f16)a1;
  }
  __syncthreads();

  // ---- density L1: feats[32x32] @ W1[32x64], ReLU -> sAct[1] ----
#pragma unroll
  for (int mt = 0; mt < 2; ++mt) {
    v16h a = frag_ld(&sAct[0][wave][mt * 16 + lc][0], 0, hi);
#pragma unroll
    for (int nt = 0; nt < 4; ++nt) {
      v16h b = frag_ld(&sW1[nt * 16 + lc][0], 0, hi);
      v8f c = {};
      c = wmma16(a, b, c);
      int col = nt * 16 + lc;
      float bias = sB1[col];
#pragma unroll
      for (int j = 0; j < 8; ++j) {
        int r = mt * 16 + (hi ? j + 8 : j);
        float v = c[j] + bias;
        sAct[1][wave][r][col] = (f16)(v > 0.f ? v : 0.f);
      }
    }
  }
  __syncthreads();

  // ---- density L2: h[32x64] @ W2[64x16] -> log_sigma; sigma out ----
  {
    v16h b0 = frag_ld(&sW2[lc][0], 0, hi);
    v16h b1 = frag_ld(&sW2[lc][0], 32, hi);
#pragma unroll
    for (int mt = 0; mt < 2; ++mt) {
      v8f c = {};
      c = wmma16(frag_ld(&sAct[1][wave][mt * 16 + lc][0], 0, hi), b0, c);
      c = wmma16(frag_ld(&sAct[1][wave][mt * 16 + lc][0], 32, hi), b1, c);
      float bias = sB2[lc];
#pragma unroll
      for (int j = 0; j < 8; ++j) {
        int r = mt * 16 + (hi ? j + 8 : j);
        float ls = c[j] + bias;
        sAct[0][wave][r][lc] = (f16)ls;  // cols 0..15 of color input
        if (lc == 0) {
          int gp = blockbase + wave * 32 + r;
          if (gp < P) outSigma[gp] = expf(ls) * sMask[wave * 32 + r];
        }
      }
    }
  }
  __syncthreads();

  // ---- build color-net input cols 16..63 (dir + positional encoding) ----
  {
    float r0 = rdir[p * 3 + 0], r1 = rdir[p * 3 + 1], r2 = rdir[p * 3 + 2];
    f16* crow = &sAct[0][wave][lane][0];
    crow[16] = (f16)r0; crow[17] = (f16)r1; crow[18] = (f16)r2;
    const float twoPi = 6.283185307179586f;
#pragma unroll
    for (int ci = 0; ci < 3; ++ci) {
      float rv = (ci == 0) ? r0 : (ci == 1) ? r1 : r2;
#pragma unroll
      for (int l = 0; l < 4; ++l) {
        float xi = rv * twoPi * (float)(1 << l);
        crow[19 + ci * 8 + l]     = (f16)sinf(xi);
        crow[19 + ci * 8 + 4 + l] = (f16)cosf(xi);
      }
    }
#pragma unroll
    for (int k = 43; k < 64; ++k) crow[k] = (f16)0.f;
  }
  __syncthreads();

  // ---- color L1: in[32x64] @ cW1[64x64], ReLU -> sAct[1] ----
#pragma unroll
  for (int mt = 0; mt < 2; ++mt) {
    v16h a0 = frag_ld(&sAct[0][wave][mt * 16 + lc][0], 0, hi);
    v16h a1 = frag_ld(&sAct[0][wave][mt * 16 + lc][0], 32, hi);
#pragma unroll
    for (int nt = 0; nt < 4; ++nt) {
      v8f c = {};
      c = wmma16(a0, frag_ld(&sC1[nt * 16 + lc][0], 0, hi), c);
      c = wmma16(a1, frag_ld(&sC1[nt * 16 + lc][0], 32, hi), c);
      int col = nt * 16 + lc;
      float bias = sCB1[col];
#pragma unroll
      for (int j = 0; j < 8; ++j) {
        int r = mt * 16 + (hi ? j + 8 : j);
        float v = c[j] + bias;
        sAct[1][wave][r][col] = (f16)(v > 0.f ? v : 0.f);
      }
    }
  }
  __syncthreads();

  // ---- color L2: [32x64] @ cW2[64x64], ReLU -> sAct[0] ----
#pragma unroll
  for (int mt = 0; mt < 2; ++mt) {
    v16h a0 = frag_ld(&sAct[1][wave][mt * 16 + lc][0], 0, hi);
    v16h a1 = frag_ld(&sAct[1][wave][mt * 16 + lc][0], 32, hi);
#pragma unroll
    for (int nt = 0; nt < 4; ++nt) {
      v8f c = {};
      c = wmma16(a0, frag_ld(&sC2[nt * 16 + lc][0], 0, hi), c);
      c = wmma16(a1, frag_ld(&sC2[nt * 16 + lc][0], 32, hi), c);
      int col = nt * 16 + lc;
      float bias = sCB2[col];
#pragma unroll
      for (int j = 0; j < 8; ++j) {
        int r = mt * 16 + (hi ? j + 8 : j);
        float v = c[j] + bias;
        sAct[0][wave][r][col] = (f16)(v > 0.f ? v : 0.f);
      }
    }
  }
  __syncthreads();

  // ---- color L3: [32x64] @ cW3[64x3 pad16], sigmoid * mask -> global ----
  {
    v16h b0 = frag_ld(&sC3[lc][0], 0, hi);
    v16h b1 = frag_ld(&sC3[lc][0], 32, hi);
#pragma unroll
    for (int mt = 0; mt < 2; ++mt) {
      v8f c = {};
      c = wmma16(frag_ld(&sAct[0][wave][mt * 16 + lc][0], 0, hi), b0, c);
      c = wmma16(frag_ld(&sAct[0][wave][mt * 16 + lc][0], 32, hi), b1, c);
      if (lc < 3) {
        float bias = sCB3[lc];
#pragma unroll
        for (int j = 0; j < 8; ++j) {
          int r = mt * 16 + (hi ? j + 8 : j);
          int gp = blockbase + wave * 32 + r;
          float v = 1.f / (1.f + expf(-(c[j] + bias)));
          if (gp < P) outColor[gp * 3 + lc] = v * sMask[wave * 32 + r];
        }
      }
    }
  }
}

extern "C" void kernel_launch(void* const* d_in, const int* in_sizes, int n_in,
                              void* d_out, int out_size, void* d_ws,
                              size_t ws_size, hipStream_t stream) {
  (void)n_in; (void)d_ws; (void)ws_size; (void)out_size;
  const float* x      = (const float*)d_in[0];
  const float* rdir   = (const float*)d_in[1];
  const unsigned char* occ = (const unsigned char*)d_in[2];
  const float* tables = (const float*)d_in[3];
  const float* dW1 = (const float*)d_in[4];
  const float* db1 = (const float*)d_in[5];
  const float* dW2 = (const float*)d_in[6];
  const float* db2 = (const float*)d_in[7];
  const float* cW1 = (const float*)d_in[8];
  const float* cb1 = (const float*)d_in[9];
  const float* cW2 = (const float*)d_in[10];
  const float* cb2 = (const float*)d_in[11];
  const float* cW3 = (const float*)d_in[12];
  const float* cb3 = (const float*)d_in[13];

  const int P = in_sizes[0] / 3;
  float* out = (float*)d_out;
  float* outColor = out;                    // [P,3]
  float* outSigma = out + (size_t)P * 3;    // [P,1]
  float* outMask  = out + (size_t)P * 4;    // [P,1]

  const int blocks = (P + 127) / 128;
  ngp_fused_kernel<<<blocks, 128, 0, stream>>>(
      x, rdir, occ, tables, dW1, db1, dW2, db2, cW1, cb1, cW2, cb2, cW3, cb3,
      outColor, outSigma, outMask, P);
}
